// cls_predictor_69526930588021
// MI455X (gfx1250) — matrix operand order repaired and checked by reading, not compiled, not run on hardware
//
#include <hip/hip_runtime.h>

typedef __attribute__((ext_vector_type(16))) _Float16 v16h;
typedef __attribute__((ext_vector_type(8)))  _Float16 v8h;
typedef __attribute__((ext_vector_type(8)))  float    v8f;

static constexpr int kB   = 32;
static constexpr int kN   = 16384;
static constexpr int kNP1 = 128;   // SA1 centroids
static constexpr int kNS1 = 32;    // SA1 samples
static constexpr int kNP2 = 32;    // SA2 centroids
static constexpr int kNS2 = 64;    // SA2 samples

#define CAT16(lo, hi) __builtin_shufflevector(lo, hi, 0, 1, 2, 3, 4, 5, 6, 7, \
                                              8, 9, 10, 11, 12, 13, 14, 15)

// ===================== templated wave32 MLP layer (WMMA) ======================
// H = relu(X @ W + b)  [optionally fused with column-max over rows -> outRow]
// Xs : LDS f16 [R][CINPAD]  (CINPAD mult of 32, 16B-aligned rows)
// Wg : global f32 [CIN][COUT];  bg: global f32 [COUT]
// Wt : LDS f16 scratch [32][CINPAD]  (weight chunk, TRANSPOSED: [col][K])
// Hs : LDS f16 [R][COUT]   (written unless FINAL)
// A frag: lane&15 = row, elems 0-7 = K 8*(lane>>4)+0..7, 8-15 = K 16+8*(lane>>4)+0..7; 2x16B per lane.
// B frag: lane&15 = col,  same K order as A; 2x16B per lane like A (transposed W).
// D frag: elem v -> row = mt + (lane>>4)*8 + v, col = nt + (lane&15).
template <int R, int CIN, int CINPAD, int COUT, bool FINAL>
__device__ __forceinline__ void wave_mlp(const _Float16* __restrict__ Xs,
                                         const float* __restrict__ Wg,
                                         const float* __restrict__ bg,
                                         _Float16* __restrict__ Wt,
                                         _Float16* __restrict__ Hs,
                                         float* __restrict__ outRow, int lane) {
  constexpr int KT = CINPAD / 32;
  const int col = lane & 15, half = lane >> 4;
  for (int nc = 0; nc < COUT; nc += 32) {
    // Load/convert W[0:CIN][nc:nc+32] into Wt transposed; global reads coalesced.
    for (int i = lane; i < CINPAD * 32; i += 32) {
      const int r = i >> 5, c = i & 31;  // r = K index, c = lane = column
      Wt[c * CINPAD + r] = (_Float16)((r < CIN) ? Wg[r * COUT + nc + c] : 0.0f);
    }
    float cmax[2] = {0.0f, 0.0f};  // FINAL: per-nt column max (relu >= 0)
    for (int mt = 0; mt < R; mt += 16) {
      v16h afr[KT];
#pragma unroll
      for (int k = 0; k < KT; ++k) {  // two ds_load_b128 per fragment
        const _Float16* xr = Xs + (mt + col) * CINPAD + k * 32 + half * 8;
        afr[k] = CAT16(*(const v8h*)xr, *(const v8h*)(xr + 16));
      }
#pragma unroll
      for (int ni = 0; ni < 2; ++ni) {
        const int nt = ni * 16;
        v8f acc = {};
#pragma unroll
        for (int k = 0; k < KT; ++k) {
          const _Float16* wr = Wt + (nt + col) * CINPAD + k * 32 + half * 8;
          v16h bfr = CAT16(*(const v8h*)wr, *(const v8h*)(wr + 16));
          acc = __builtin_amdgcn_wmma_f32_16x16x32_f16(false, afr[k], false, bfr,
                                                       (short)0, acc, false, false);
        }
        const float bb = bg[nc + nt + col];
        if (FINAL) {
#pragma unroll
          for (int v = 0; v < 8; ++v)
            cmax[ni] = fmaxf(cmax[ni], fmaxf(acc[v] + bb, 0.0f));
        } else {
#pragma unroll
          for (int v = 0; v < 8; ++v) {
            const float x = fmaxf(acc[v] + bb, 0.0f);
            Hs[(mt + half * 8 + v) * COUT + (nc + nt + col)] = (_Float16)x;
          }
        }
      }
    }
    if (FINAL) {
#pragma unroll
      for (int ni = 0; ni < 2; ++ni) {
        float cm = fmaxf(cmax[ni], __shfl_xor(cmax[ni], 16, 32));
        if (lane < 16) outRow[nc + ni * 16 + col] = cm;
      }
    }
  }
}

// ============================ FPS (stage 1, N=16384) ==========================
__global__ void fps1_kernel(const float* __restrict__ xyz, float* __restrict__ dist,
                            float* __restrict__ nxyz) {
  constexpr int NT = 256;
  __shared__ float sval[NT];
  __shared__ int   sidx[NT];
  __shared__ int   sFps[kNP1];
  __shared__ int   sFar;
  const int b = blockIdx.x, t = threadIdx.x;
  const float* X = xyz + (size_t)b * kN * 3;
  float* D = dist + (size_t)b * kN;
  for (int i = t; i < kN; i += NT) D[i] = 1e10f;
  if (t == 0) sFar = 0;
  __syncthreads();
  for (int it = 0; it < kNP1; ++it) {
    const int far = sFar;
    if (t == 0) sFps[it] = far;
    const float cx = X[far * 3], cy = X[far * 3 + 1], cz = X[far * 3 + 2];
    float bv = -1.0f; int bi = 0;
    for (int i = t; i < kN; i += NT) {
      float dx = X[i * 3] - cx, dy = X[i * 3 + 1] - cy, dz = X[i * 3 + 2] - cz;
      float dd = fminf(D[i], dx * dx + dy * dy + dz * dz);
      D[i] = dd;
      if (dd > bv) { bv = dd; bi = i; }
    }
    sval[t] = bv; sidx[t] = bi;
    __syncthreads();
    for (int off = NT / 2; off > 0; off >>= 1) {
      if (t < off) {
        float v2 = sval[t + off]; int i2 = sidx[t + off];
        if (v2 > sval[t] || (v2 == sval[t] && i2 < sidx[t])) { sval[t] = v2; sidx[t] = i2; }
      }
      __syncthreads();
    }
    if (t == 0) sFar = sidx[0];
    __syncthreads();
  }
  if (t < kNP1) {
    const int j = sFps[t];
    nxyz[((size_t)b * kNP1 + t) * 3 + 0] = X[j * 3 + 0];
    nxyz[((size_t)b * kNP1 + t) * 3 + 1] = X[j * 3 + 1];
    nxyz[((size_t)b * kNP1 + t) * 3 + 2] = X[j * 3 + 2];
  }
}

// ============================ FPS (stage 2, 128 pts) ==========================
__global__ void fps2_kernel(const float* __restrict__ xyz1, float* __restrict__ xyz2) {
  __shared__ float D[kNP1];
  __shared__ float sval[kNP1];
  __shared__ int   sidx[kNP1];
  __shared__ int   sFps[kNP2];
  __shared__ int   sFar;
  const int b = blockIdx.x, t = threadIdx.x;  // 128 threads
  const float* X = xyz1 + (size_t)b * kNP1 * 3;
  D[t] = 1e10f;
  if (t == 0) sFar = 0;
  __syncthreads();
  for (int it = 0; it < kNP2; ++it) {
    const int far = sFar;
    if (t == 0) sFps[it] = far;
    const float cx = X[far * 3], cy = X[far * 3 + 1], cz = X[far * 3 + 2];
    float dx = X[t * 3] - cx, dy = X[t * 3 + 1] - cy, dz = X[t * 3 + 2] - cz;
    float dd = fminf(D[t], dx * dx + dy * dy + dz * dz);
    D[t] = dd;
    sval[t] = dd; sidx[t] = t;
    __syncthreads();
    for (int off = kNP1 / 2; off > 0; off >>= 1) {
      if (t < off) {
        float v2 = sval[t + off]; int i2 = sidx[t + off];
        if (v2 > sval[t] || (v2 == sval[t] && i2 < sidx[t])) { sval[t] = v2; sidx[t] = i2; }
      }
      __syncthreads();
    }
    if (t == 0) sFar = sidx[0];
    __syncthreads();
  }
  if (t < kNP2) {
    const int j = sFps[t];
    xyz2[((size_t)b * kNP2 + t) * 3 + 0] = X[j * 3 + 0];
    xyz2[((size_t)b * kNP2 + t) * 3 + 1] = X[j * 3 + 1];
    xyz2[((size_t)b * kNP2 + t) * 3 + 2] = X[j * 3 + 2];
  }
}

// ============ Ball query: first nsample in-radius indices (ascending) =========
__global__ void ballq_kernel(const float* __restrict__ pts, int npts,
                             const float* __restrict__ cen, int nper, int total,
                             int nsample, float r2, int* __restrict__ idx) {
  const int t = blockIdx.x * blockDim.x + threadIdx.x;
  if (t >= total) return;
  const int b = t / nper;
  const float* P = pts + (size_t)b * npts * 3;
  const float cx = cen[t * 3], cy = cen[t * 3 + 1], cz = cen[t * 3 + 2];
  int* out = idx + (size_t)t * nsample;
  int cnt = 0, first = 0;
  for (int i = 0; i < npts && cnt < nsample; ++i) {
    float dx = P[i * 3] - cx, dy = P[i * 3 + 1] - cy, dz = P[i * 3 + 2] - cz;
    if (dx * dx + dy * dy + dz * dz <= r2) {
      if (cnt == 0) first = i;
      out[cnt++] = i;
    }
  }
  if (cnt == 0) { out[0] = 0; cnt = 1; first = 0; }
  for (; cnt < nsample; ++cnt) out[cnt] = first;
}

// =================== SA1: 4096 groups, rows=32, 3->64->64->128 ================
__global__ void __launch_bounds__(32) sa1_mlp_kernel(
    const float* __restrict__ xyz, const float* __restrict__ nxyz,
    const int* __restrict__ idx,
    const float* w1, const float* b1, const float* w2, const float* b2,
    const float* w3, const float* b3, float* __restrict__ f1) {
  __shared__ alignas(16) _Float16 bufA[32 * 64];
  __shared__ alignas(16) _Float16 bufB[32 * 64];
  __shared__ alignas(16) _Float16 Wt[32 * 64];
  const int g = blockIdx.x;            // b*128 + s
  const int b = g >> 7, s = g & 127;
  const int lane = threadIdx.x;
  {
    const int j = idx[((size_t)b * kNP1 + s) * kNS1 + lane];
    const float cx = nxyz[((size_t)b * kNP1 + s) * 3 + 0];
    const float cy = nxyz[((size_t)b * kNP1 + s) * 3 + 1];
    const float cz = nxyz[((size_t)b * kNP1 + s) * 3 + 2];
    const float* p = xyz + ((size_t)b * kN + j) * 3;
    _Float16* xr = bufA + lane * 32;
    xr[0] = (_Float16)((p[0] - cx) * 5.0f);   // /radius 0.2
    xr[1] = (_Float16)((p[1] - cy) * 5.0f);
    xr[2] = (_Float16)((p[2] - cz) * 5.0f);
    for (int c = 3; c < 32; ++c) xr[c] = (_Float16)0.0f;
  }
  wave_mlp<32,  3, 32,  64, false>(bufA, w1, b1, Wt, bufB, nullptr, lane);
  wave_mlp<32, 64, 64,  64, false>(bufB, w2, b2, Wt, bufA, nullptr, lane);
  wave_mlp<32, 64, 64, 128, true >(bufA, w3, b3, Wt, nullptr,
                                   f1 + ((size_t)b * kNP1 + s) * 128, lane);
}

// ================ SA2: 1024 groups, rows=64, 131->128->128->256 ===============
__global__ void __launch_bounds__(32) sa2_mlp_kernel(
    const float* __restrict__ xyz1, const float* __restrict__ xyz2,
    const int* __restrict__ idx, const float* __restrict__ f1,
    const float* w1, const float* b1, const float* w2, const float* b2,
    const float* w3, const float* b3, float* __restrict__ f2) {
  __shared__ alignas(16) _Float16 bufA[64 * 160];
  __shared__ alignas(16) _Float16 bufB[64 * 128];
  __shared__ alignas(16) _Float16 Wt[32 * 160];
  const int g = blockIdx.x;            // b*32 + s
  const int b = g >> 5, s = g & 31;
  const int lane = threadIdx.x;
  const float cx = xyz2[((size_t)b * kNP2 + s) * 3 + 0];
  const float cy = xyz2[((size_t)b * kNP2 + s) * 3 + 1];
  const float cz = xyz2[((size_t)b * kNP2 + s) * 3 + 2];
  for (int r = lane; r < kNS2; r += 32) {
    const int j = idx[((size_t)b * kNP2 + s) * kNS2 + r];
    const float* p = xyz1 + ((size_t)b * kNP1 + j) * 3;
    _Float16* xr = bufA + r * 160;
    xr[0] = (_Float16)((p[0] - cx) * 2.5f);   // /radius 0.4
    xr[1] = (_Float16)((p[1] - cy) * 2.5f);
    xr[2] = (_Float16)((p[2] - cz) * 2.5f);
    const float* fp = f1 + ((size_t)b * kNP1 + j) * 128;
    for (int c = 0; c < 128; ++c) xr[3 + c] = (_Float16)fp[c];
    for (int c = 131; c < 160; ++c) xr[c] = (_Float16)0.0f;
  }
  wave_mlp<64, 131, 160, 128, false>(bufA, w1, b1, Wt, bufB, nullptr, lane);
  wave_mlp<64, 128, 128, 128, false>(bufB, w2, b2, Wt, bufA, nullptr, lane);
  wave_mlp<64, 128, 128, 256, true >(bufA, w3, b3, Wt, nullptr,
                                     f2 + ((size_t)b * kNP2 + s) * 256, lane);
}

// ============== SA3 (group-all): 32 groups, rows=32, 259->128->128->256 =======
__global__ void __launch_bounds__(32) sa3_mlp_kernel(
    const float* __restrict__ xyz2, const float* __restrict__ f2,
    const float* w1, const float* b1, const float* w2, const float* b2,
    const float* w3, const float* b3, float* __restrict__ f3) {
  __shared__ alignas(16) _Float16 bufA[32 * 288];
  __shared__ alignas(16) _Float16 bufB[32 * 128];
  __shared__ alignas(16) _Float16 Wt[32 * 288];
  const int b = blockIdx.x, lane = threadIdx.x;
  {
    const float* p = xyz2 + ((size_t)b * kNP2 + lane) * 3;
    _Float16* xr = bufA + lane * 288;
    xr[0] = (_Float16)p[0]; xr[1] = (_Float16)p[1]; xr[2] = (_Float16)p[2];
    const float* fp = f2 + ((size_t)b * kNP2 + lane) * 256;
    for (int c = 0; c < 256; ++c) xr[3 + c] = (_Float16)fp[c];
    for (int c = 259; c < 288; ++c) xr[c] = (_Float16)0.0f;
  }
  wave_mlp<32, 259, 288, 128, false>(bufA, w1, b1, Wt, bufB, nullptr, lane);
  wave_mlp<32, 128, 128, 128, false>(bufB, w2, b2, Wt, bufA, nullptr, lane);
  wave_mlp<32, 128, 128, 256, true >(bufA, w3, b3, Wt, nullptr,
                                     f3 + (size_t)b * 256, lane);
}

// ======================== Head: 256 -> 128 (relu) -> 18 =======================
__global__ void head_kernel(const float* __restrict__ f3,
                            const float* w1, const float* b1,
                            const float* w2, const float* b2,
                            float* __restrict__ out) {
  __shared__ float h[128];
  const int b = blockIdx.x, t = threadIdx.x;  // 128 threads
  const float* x = f3 + (size_t)b * 256;
  float acc = b1[t];
  for (int k = 0; k < 256; ++k) acc += x[k] * w1[k * 128 + t];
  h[t] = fmaxf(acc, 0.0f);
  __syncthreads();
  if (t < 18) {
    float o = b2[t];
    for (int k = 0; k < 128; ++k) o += h[k] * w2[k * 18 + t];
    out[b * 18 + t] = o;
  }
}

// ================================== launch ===================================
extern "C" void kernel_launch(void* const* d_in, const int* in_sizes, int n_in,
                              void* d_out, int out_size, void* d_ws, size_t ws_size,
                              hipStream_t stream) {
  (void)in_sizes; (void)n_in; (void)out_size; (void)ws_size;
  const float* xyz   = (const float*)d_in[0];
  const float* sa1w[3] = {(const float*)d_in[1], (const float*)d_in[2], (const float*)d_in[3]};
  const float* sa1b[3] = {(const float*)d_in[4], (const float*)d_in[5], (const float*)d_in[6]};
  const float* sa2w[3] = {(const float*)d_in[7], (const float*)d_in[8], (const float*)d_in[9]};
  const float* sa2b[3] = {(const float*)d_in[10], (const float*)d_in[11], (const float*)d_in[12]};
  const float* sa3w[3] = {(const float*)d_in[13], (const float*)d_in[14], (const float*)d_in[15]};
  const float* sa3b[3] = {(const float*)d_in[16], (const float*)d_in[17], (const float*)d_in[18]};
  const float* hw1 = (const float*)d_in[19];
  const float* hw2 = (const float*)d_in[20];
  const float* hb1 = (const float*)d_in[21];
  const float* hb2 = (const float*)d_in[22];
  float* out = (float*)d_out;

  char* ws = (char*)d_ws;
  float* dist = (float*)ws;  ws += (size_t)kB * kN * 4;                 // 2 MB
  float* xyz1 = (float*)ws;  ws += (size_t)kB * kNP1 * 3 * 4;
  int*   idx1 = (int*)ws;    ws += (size_t)kB * kNP1 * kNS1 * 4;
  float* f1   = (float*)ws;  ws += (size_t)kB * kNP1 * 128 * 4;         // 2 MB
  float* xyz2 = (float*)ws;  ws += (size_t)kB * kNP2 * 3 * 4;
  int*   idx2 = (int*)ws;    ws += (size_t)kB * kNP2 * kNS2 * 4;
  float* f2   = (float*)ws;  ws += (size_t)kB * kNP2 * 256 * 4;         // 1 MB
  float* f3   = (float*)ws;  ws += (size_t)kB * 256 * 4;

  fps1_kernel<<<kB, 256, 0, stream>>>(xyz, dist, xyz1);

  {
    const int total = kB * kNP1;
    ballq_kernel<<<(total + 127) / 128, 128, 0, stream>>>(
        xyz, kN, xyz1, kNP1, total, kNS1, 0.2f * 0.2f, idx1);
  }

  sa1_mlp_kernel<<<kB * kNP1, 32, 0, stream>>>(
      xyz, xyz1, idx1, sa1w[0], sa1b[0], sa1w[1], sa1b[1], sa1w[2], sa1b[2], f1);

  fps2_kernel<<<kB, kNP1, 0, stream>>>(xyz1, xyz2);

  {
    const int total = kB * kNP2;
    ballq_kernel<<<(total + 127) / 128, 128, 0, stream>>>(
        xyz1, kNP1, xyz2, kNP2, total, kNS2, 0.4f * 0.4f, idx2);
  }

  sa2_mlp_kernel<<<kB * kNP2, 32, 0, stream>>>(
      xyz1, xyz2, idx2, f1, sa2w[0], sa2b[0], sa2w[1], sa2b[1], sa2w[2], sa2b[2], f2);

  sa3_mlp_kernel<<<kB, 32, 0, stream>>>(
      xyz2, f2, sa3w[0], sa3b[0], sa3w[1], sa3b[1], sa3w[2], sa3b[2], f3);

  head_kernel<<<kB, 128, 0, stream>>>(f3, hw1, hb1, hw2, hb2, out);
}
